// aformer_59407987638526
// MI455X (gfx1250) — compile-verified, matrix operand order checked by reading, not run on hardware
//
#include <hip/hip_runtime.h>
#include <math.h>

// ---------------- CDNA5 WMMA GEMM (implicit im2col, fused pre/post ops) ----------------
// Tile: 128x64 output per workgroup, 256 threads = 8 wave32 waves. Each wave owns a
// 16-row strip and all four 16-wide column fragments: per BK=32 K-step it loads ONE A
// fragment + FOUR B fragments (into distinct registers, so one dscnt wait covers all)
// and issues FOUR back-to-back v_wmma_f32_16x16x32_f16, accumulating fp32 across K.
// A is gathered (optionally im2col, template-specialized) from a float tensor with
// fused per-channel affine (BatchNorm) + tanh-GELU + scalar alpha, converted to f16 in
// LDS. B is staged transposed ([n][k]) so each lane's fragment is two aligned 16-byte
// ds_load_b128s. Fused bias / residual-add / GELU on store.

#define TBM 128
#define TBN 64
#define TBK 32
#define AST 40   // LDS row stride (halves): 80B rows, 16B-aligned sub-loads
#define BST 40
#define HEAD_D 32

typedef __attribute__((ext_vector_type(16))) _Float16 v16h;
typedef __attribute__((ext_vector_type(8)))  _Float16 v8h;
typedef __attribute__((ext_vector_type(8)))  float    v8f;

__device__ __forceinline__ float geluf(float x) {
  return 0.5f * x * (1.0f + tanhf(0.7978845608028654f * (x + 0.044715f * x * x * x)));
}

template<bool IM2COL>
__global__ __launch_bounds__(256)
void wmma_gemm_kernel(const float* __restrict__ X, int Lx, int lda, int aOff,
                      int Cin, int taps, int pad,
                      const float* __restrict__ W, int ldb, int bTrans,
                      const float* __restrict__ scale, const float* __restrict__ shift,
                      float affMul, int preGelu, float alpha,
                      const float* __restrict__ bias,
                      const float* __restrict__ base, int ldBase,
                      float* __restrict__ C, int ldc,
                      int M, int N, int K, int postGelu)
{
  __shared__ __align__(16) _Float16 sA[TBM * AST];
  __shared__ __align__(16) _Float16 sB[TBN * BST];

  const int tid = threadIdx.x;
  const int m0  = blockIdx.y * TBM;
  const int n0  = blockIdx.x * TBN;

  const int w   = tid >> 5;          // wave id 0..7
  const int ln  = tid & 15;          // lane % 16
  const int hi  = (tid >> 4) & 1;    // lane half (0: lanes 0-15, 1: lanes 16-31)
  const int fm  = w * 16;            // this wave's 16-row strip

  v8f acc[4];
  #pragma unroll
  for (int f = 0; f < 4; ++f) acc[f] = v8f{0.f,0.f,0.f,0.f,0.f,0.f,0.f,0.f};

  const int kSteps = (K + TBK - 1) / TBK;
  for (int ks = 0; ks < kSteps; ++ks) {
    const int k0 = ks * TBK;

    // prefetch next B tile (global_prefetch_b8 on gfx1250)
    if (ks + 1 < kSteps && tid < TBK) {
      int kk = k0 + TBK + tid;
      if (kk < K) {
        const float* p = bTrans ? &W[(size_t)(n0) * ldb + kk]
                                : &W[(size_t)kk * ldb + n0];
        __builtin_prefetch(p, 0, 1);
      }
    }

    // ---- stage A: 128 rows x 32 halves; thread covers 16 contiguous cols of one row ----
    {
      const int r     = tid >> 1;
      const int cbase = (tid & 1) * 16;
      const int m     = m0 + r;
      #pragma unroll
      for (int j = 0; j < 16; ++j) {
        const int kk = k0 + cbase + j;
        float v = 0.f;
        if (kk < K && m < M) {
          if (IM2COL) {
            const int tap = kk / Cin;
            const int c   = kk - tap * Cin;
            const int row = m + tap - pad;
            if (row >= 0 && row < Lx) {
              v = X[(size_t)row * lda + aOff + c];
              if (scale)   v = v * (scale[c] * affMul) + shift[c];
              if (preGelu) v = geluf(v);
              v *= alpha;
            }
          } else {
            v = X[(size_t)m * lda + aOff + kk];
            if (scale)   v = v * (scale[kk] * affMul) + shift[kk];
            if (preGelu) v = geluf(v);
            v *= alpha;
          }
        }
        sA[r * AST + cbase + j] = (_Float16)v;
      }
    }
    // ---- stage B transposed into sB[n][k] (64 x 32) ----
    if (!bTrans) {
      // W row-major [K][N]: coalesced reads over n, scatter into Bt
      const int lin  = tid * 8;
      const int kk_l = lin >> 6;       // / TBN
      const int nn   = lin & 63;
      const int kk   = k0 + kk_l;
      #pragma unroll
      for (int j = 0; j < 8; ++j) {
        const int n = n0 + nn + j;
        float v = (kk < K && n < N) ? W[(size_t)kk * ldb + n] : 0.f;
        sB[(nn + j) * BST + kk_l] = (_Float16)v;
      }
    } else {
      // W row-major [N][K]: contiguous reads over k
      const int lin = tid * 8;
      const int nn  = lin >> 5;        // / TBK
      const int kb  = lin & 31;
      const int n   = n0 + nn;
      #pragma unroll
      for (int j = 0; j < 8; ++j) {
        const int kk = k0 + kb + j;
        float v = (kk < K && n < N) ? W[(size_t)n * ldb + kk] : 0.f;
        sB[nn * BST + kb + j] = (_Float16)v;
      }
    }
    __syncthreads();

    // ---- fragments (ISA 7.12.2 f16 A 16x32 layout): halves 0..7 -> K=hi*8..,
    //      halves 8..15 -> K=16+hi*8..; B mirrors A with n on lanes ----
    v16h a;
    {
      const v8h lo = *(const v8h*)&sA[(fm + ln) * AST + hi * 8];
      const v8h hv = *(const v8h*)&sA[(fm + ln) * AST + 16 + hi * 8];
      #pragma unroll
      for (int i = 0; i < 8; ++i) { a[i] = lo[i]; a[8 + i] = hv[i]; }
    }
    // load ALL four B fragments first (distinct registers -> one dscnt wait),
    // then issue the four WMMAs back-to-back
    v16h barr[4];
    #pragma unroll
    for (int f = 0; f < 4; ++f) {
      const v8h lo = *(const v8h*)&sB[(f * 16 + ln) * BST + hi * 8];
      const v8h hv = *(const v8h*)&sB[(f * 16 + ln) * BST + 16 + hi * 8];
      #pragma unroll
      for (int i = 0; i < 8; ++i) { barr[f][i] = lo[i]; barr[f][8 + i] = hv[i]; }
    }
    #pragma unroll
    for (int f = 0; f < 4; ++f) {
      acc[f] = __builtin_amdgcn_wmma_f32_16x16x32_f16(false, a, false, barr[f], (short)0, acc[f], false, false);
    }
    __syncthreads();
  }

  // ---- writeout: C/D layout VGPR r -> M = r + hi*8, N = ln ----
  #pragma unroll
  for (int r = 0; r < 8; ++r) {
    const int row = m0 + fm + r + hi * 8;
    if (row >= M) continue;
    #pragma unroll
    for (int f = 0; f < 4; ++f) {
      const int col = n0 + f * 16 + ln;
      if (col < N) {
        float v = acc[f][r];
        if (bias)     v += bias[col];
        if (base)     v += base[(size_t)row * ldBase + col];
        if (postGelu) v = geluf(v);
        C[(size_t)row * ldc + col] = v;
      }
    }
  }
}

// ---------------- small VALU kernels ----------------

// softmax_pool: out[l,c] = x[2l,c]*w0 + x[2l+1,c]*(1-w0), w0 = sigmoid(logit0-logit1)
__global__ void pool_kernel(const float* __restrict__ x, const float* __restrict__ logits,
                            float* __restrict__ out, int Lout, int C) {
  int idx = blockIdx.x * blockDim.x + threadIdx.x;
  if (idx >= Lout * C) return;
  int c = idx % C, l = idx / C;
  float l0 = logits[(size_t)(2 * l) * C + c];
  float l1 = logits[(size_t)(2 * l + 1) * C + c];
  float w0 = 1.f / (1.f + expf(l1 - l0));
  float x0 = x[(size_t)(2 * l) * C + c];
  float x1 = x[(size_t)(2 * l + 1) * C + c];
  out[idx] = x0 * w0 + x1 * (1.f - w0);
}

// LayerNorm over C=128 (one row per 128-thread block; wave32 shuffle + LDS reduce)
__global__ __launch_bounds__(128)
void ln_kernel(const float* __restrict__ x, const float* __restrict__ g,
               const float* __restrict__ b, float* __restrict__ y, int C) {
  int row = blockIdx.x, c = threadIdx.x;
  float v = x[(size_t)row * C + c];
  float sum = v, sq = v * v;
  #pragma unroll
  for (int o = 16; o > 0; o >>= 1) { sum += __shfl_xor(sum, o); sq += __shfl_xor(sq, o); }
  __shared__ float s1[4], s2[4];
  if ((c & 31) == 0) { s1[c >> 5] = sum; s2[c >> 5] = sq; }
  __syncthreads();
  sum = s1[0] + s1[1] + s1[2] + s1[3];
  sq  = s2[0] + s2[1] + s2[2] + s2[3];
  float mu  = sum / (float)C;
  float var = sq / (float)C - mu * mu;
  float r   = rsqrtf(var + 1e-6f);
  y[(size_t)row * C + c] = (v - mu) * r * g[c] + b[c];
}

// rotary in-place on [L, H, 32]
__global__ void rotary_kernel(float* __restrict__ x, int L, int H, int Dh) {
  int idx = blockIdx.x * blockDim.x + threadIdx.x;
  int half = Dh / 2;
  if (idx >= L * H * half) return;
  int i = idx % half;
  int h = (idx / half) % H;
  int l = idx / (half * H);
  float ang = (float)l * __powf(10000.f, -2.f * (float)i / (float)Dh);
  float s = __sinf(ang), co = __cosf(ang);
  float* p = x + ((size_t)l * H + h) * Dh;
  float x1 = p[2 * i], x2 = p[2 * i + 1];
  p[2 * i]     = x1 * co - x2 * s;
  p[2 * i + 1] = x1 * s  + x2 * co;
}

// diag[l,h] = 0.5 * sum_d (x*dn)^2
__global__ void diag_kernel(const float* __restrict__ q, float* __restrict__ diag,
                            int L, int H, float dn) {
  int idx = blockIdx.x * blockDim.x + threadIdx.x;
  if (idx >= L * H) return;
  int h = idx % H, l = idx / H;
  const float* p = q + ((size_t)l * H + h) * HEAD_D;
  float s = 0.f;
  #pragma unroll
  for (int d = 0; d < HEAD_D; ++d) { float v = p[d] * dn; s += v * v; }
  diag[idx] = 0.5f * s;
}

// per-row max over last dim (rows = H*L, width Mrf)
__global__ void rowmax_kernel(const float* __restrict__ dash, float* __restrict__ out,
                              int rows, int Mrf) {
  int r = blockIdx.x * blockDim.x + threadIdx.x;
  if (r >= rows) return;
  const float* p = dash + (size_t)r * Mrf;
  float m = -1e30f;
  for (int i = 0; i < Mrf; ++i) m = fmaxf(m, p[i]);
  out[r] = m;
}

// per-head max over L*Mrf (one block per head)
__global__ __launch_bounds__(256)
void headmax_kernel(const float* __restrict__ dash, float* __restrict__ out, int perHead) {
  int h = blockIdx.x;
  const float* p = dash + (size_t)h * perHead;
  float m = -1e30f;
  for (int i = threadIdx.x; i < perHead; i += 256) m = fmaxf(m, p[i]);
  #pragma unroll
  for (int o = 16; o > 0; o >>= 1) m = fmaxf(m, __shfl_xor(m, o));
  __shared__ float s[8];
  if ((threadIdx.x & 31) == 0) s[threadIdx.x >> 5] = m;
  __syncthreads();
  if (threadIdx.x == 0) {
    float r = s[0];
    for (int i = 1; i < 8; ++i) r = fmaxf(r, s[i]);
    out[h] = r;
  }
}

// phi = ratio * (exp(dash - diag - m) + eps), in place; dash layout [H][L][Mrf]
__global__ void phi_kernel(float* __restrict__ dash, const float* __restrict__ diag,
                           const float* __restrict__ mx, int L, int H, int Mrf,
                           int isQuery, float ratio, float eps) {
  int idx = blockIdx.x * blockDim.x + threadIdx.x;
  if (idx >= H * L * Mrf) return;
  int l = (idx / Mrf) % L;
  int h = idx / (Mrf * L);
  float mv = isQuery ? mx[(size_t)h * L + l] : mx[h];
  float d  = diag[(size_t)l * H + h];
  dash[idx] = ratio * (expf(dash[idx] - d - mv) + eps);
}

// kv[h,m,d] = sum_l kp[h,l,m]*v[l,h*32+d]; ksum[h,m] = sum_l kp[h,l,m]
__global__ __launch_bounds__(256)
void kv_kernel(const float* __restrict__ kp, const float* __restrict__ v,
               float* __restrict__ kv, float* __restrict__ ksum,
               int L, int H, int Mrf) {
  int idx = blockIdx.x * blockDim.x + threadIdx.x;
  if (idx >= H * Mrf) return;
  int m = idx % Mrf, h = idx / Mrf;
  float acc[HEAD_D];
  #pragma unroll
  for (int d = 0; d < HEAD_D; ++d) acc[d] = 0.f;
  float s = 0.f;
  const float* kph = kp + (size_t)h * L * Mrf;
  for (int l = 0; l < L; ++l) {
    float kval = kph[(size_t)l * Mrf + m];
    s += kval;
    const float* vp = v + (size_t)l * H * HEAD_D + h * HEAD_D;
    #pragma unroll
    for (int d = 0; d < HEAD_D; ++d) acc[d] += kval * vp[d];
  }
  ksum[idx] = s;
  float* o = kv + (size_t)idx * HEAD_D;
  #pragma unroll
  for (int d = 0; d < HEAD_D; ++d) o[d] = acc[d];
}

// denom[l,h] = qp[h,l,:] . ksum[h,:] + stab; att[l,h*32+d] /= denom
__global__ void denomdiv_kernel(const float* __restrict__ qp, const float* __restrict__ ksum,
                                float* __restrict__ att, int L, int H, int Mrf, float stab) {
  int idx = blockIdx.x * blockDim.x + threadIdx.x;
  if (idx >= L * H) return;
  int h = idx % H, l = idx / H;
  const float* qph = qp + ((size_t)h * L + l) * Mrf;
  const float* ks  = ksum + (size_t)h * Mrf;
  float s = stab;
  for (int m = 0; m < Mrf; ++m) s += qph[m] * ks[m];
  float inv = 1.f / s;
  float* a = att + (size_t)l * H * HEAD_D + h * HEAD_D;
  #pragma unroll
  for (int d = 0; d < HEAD_D; ++d) a[d] *= inv;
}

// h += sinusoidal PE
__global__ void addpe_kernel(float* __restrict__ h, int L, int D) {
  int idx = blockIdx.x * blockDim.x + threadIdx.x;
  if (idx >= L * D) return;
  int c = idx % D, l = idx / D;
  int i = c >> 1;
  float ang = (float)l * __powf(10000.f, -2.f * (float)i / (float)D);
  h[idx] += (c & 1) ? __cosf(ang) : __sinf(ang);
}

// concat [x | tf_tiled | TSS | exon] -> [L, C1+C2+2]
__global__ void concat_kernel(const float* __restrict__ x, int C1,
                              const float* __restrict__ tfp, int C2,
                              const float* __restrict__ tss, const float* __restrict__ exo,
                              float* __restrict__ out, int L) {
  int wdt = C1 + C2 + 2;
  int idx = blockIdx.x * blockDim.x + threadIdx.x;
  if (idx >= L * wdt) return;
  int c = idx % wdt, l = idx / wdt;
  float v;
  if (c < C1)            v = x[(size_t)l * C1 + c];
  else if (c < C1 + C2)  v = tfp[c - C1];
  else if (c == C1 + C2) v = tss[l];
  else                   v = exo[l];
  out[idx] = v;
}

// tf MLP (1x128): out = gelu(tf@w1+b1)@w2+b2, single 128-thread block
__global__ __launch_bounds__(128)
void tf_kernel(const float* __restrict__ tf, const float* __restrict__ w1,
               const float* __restrict__ b1, const float* __restrict__ w2,
               const float* __restrict__ b2, float* __restrict__ out) {
  __shared__ float sx[128], st[128];
  int j = threadIdx.x;
  sx[j] = tf[j];
  __syncthreads();
  float a = b1[j];
  for (int i = 0; i < 128; ++i) a += sx[i] * w1[i * 128 + j];
  st[j] = geluf(a);
  __syncthreads();
  float o = b2[j];
  for (int i = 0; i < 128; ++i) o += st[i] * w2[i * 128 + j];
  out[j] = o;
}

// head: out[l] = softplus(gelu(h16[l]@w1+b1)@w2+b2)
__global__ void head_kernel(const float* __restrict__ h16, const float* __restrict__ w1,
                            const float* __restrict__ b1, const float* __restrict__ w2,
                            const float* __restrict__ b2, float* __restrict__ out, int L) {
  int l = blockIdx.x * blockDim.x + threadIdx.x;
  if (l >= L) return;
  const float* x = h16 + (size_t)l * 16;
  float t[8];
  #pragma unroll
  for (int j = 0; j < 8; ++j) {
    float a = b1[j];
    #pragma unroll
    for (int i = 0; i < 16; ++i) a += x[i] * w1[i * 8 + j];
    t[j] = geluf(a);
  }
  float o = b2[0];
  #pragma unroll
  for (int j = 0; j < 8; ++j) o += t[j] * w2[j];
  out[l] = (o > 20.f) ? o : log1pf(expf(o));
}

// ---------------- host orchestration ----------------

static void launch_gemm(hipStream_t st,
                        const float* X, int Lx, int lda, int aOff, int Cin, int taps, int pad,
                        const float* W, int ldb, int bT,
                        const float* scale, const float* shift, float affMul, int preG, float alpha,
                        const float* bias, const float* base, int ldBase,
                        float* C, int ldc, int M, int N, int K, int postG) {
  dim3 grid((N + TBN - 1) / TBN, (M + TBM - 1) / TBM);
  if (taps > 1) {
    wmma_gemm_kernel<true><<<grid, 256, 0, st>>>(X, Lx, lda, aOff, Cin, taps, pad, W, ldb, bT,
                                                 scale, shift, affMul, preG, alpha,
                                                 bias, base, ldBase, C, ldc, M, N, K, postG);
  } else {
    wmma_gemm_kernel<false><<<grid, 256, 0, st>>>(X, Lx, lda, aOff, Cin, taps, pad, W, ldb, bT,
                                                  scale, shift, affMul, preG, alpha,
                                                  bias, base, ldBase, C, ldc, M, N, K, postG);
  }
}

extern "C" void kernel_launch(void* const* d_in, const int* in_sizes, int n_in,
                              void* d_out, int out_size, void* d_ws, size_t ws_size,
                              hipStream_t stream) {
  (void)in_sizes; (void)n_in; (void)out_size; (void)ws_size;

  const float BN_RS = 0.999500374688f;       // rsqrt(1 + 1e-3)
  const float DN    = 0.4204482076268573f;   // 32^(-1/4)
  const float RATIO = 0.0625f;               // 1/sqrt(256)

  // ---- inputs ----
  const float* sequence = (const float*)d_in[0];   // (65536, 4)
  const float* TSSs     = (const float*)d_in[1];   // (512, 1)
  const float* exons    = (const float*)d_in[2];   // (512, 1)
  const float* tf_in    = (const float*)d_in[3];   // (128,)
  /* d_in[4] = atac : unused by reference forward */

  // ---- params (make_params insertion order) ----
  int pi = 5;
  auto P = [&]() -> const float* { return (const float*)d_in[pi++]; };
  const float* stem_conv_k = P(); const float* stem_conv_b = P();
  const float* stem_res_g = P();  const float* stem_res_b = P();
  const float* stem_res_k = P();  const float* stem_res_kb = P();
  const float* stem_pool_w = P();
  const int FILT[6] = {384, 448, 512, 576, 640, 768};
  const float *tg1[6], *tb1[6], *tk1[6], *tkb1[6], *tg2[6], *tb2[6], *tk2[6], *tkb2[6], *tpw[6];
  for (int i = 0; i < 6; ++i) {
    tg1[i] = P(); tb1[i] = P(); tk1[i] = P(); tkb1[i] = P();
    tg2[i] = P(); tb2[i] = P(); tk2[i] = P(); tkb2[i] = P(); tpw[i] = P();
  }
  const float* tf_w1 = P(); const float* tf_b1 = P();
  const float* tf_w2 = P(); const float* tf_b2 = P();
  const float* mix_k = P(); const float* mix_b = P();
  const float *ln1g[4], *ln1b[4], *wq[4], *wk[4], *wv[4], *wo[4], *wob[4];
  const float *ln2g[4], *ln2b[4], *f1[4], *f1b[4], *f2[4], *f2b[4];
  for (int i = 0; i < 4; ++i) {
    ln1g[i] = P(); ln1b[i] = P(); wq[i] = P(); wk[i] = P(); wv[i] = P();
    wo[i] = P(); wob[i] = P(); ln2g[i] = P(); ln2b[i] = P();
    f1[i] = P(); f1b[i] = P(); f2[i] = P(); f2b[i] = P();
  }
  const float* final_ln_g = P(); const float* final_ln_b = P();
  const float* proj = P();                        // (256, 32)
  const float* fp_g = P(); const float* fp_b = P();
  const float* fp_k = P(); const float* fp_kb = P();
  const float* head_w1 = P(); const float* head_b1 = P();
  const float* head_w2 = P(); const float* head_b2 = P();

  // ---- workspace bump allocator ----
  char* ws = (char*)d_ws;
  size_t off = 0;
  auto alloc = [&](size_t nfloats) -> float* {
    float* p = (float*)(ws + off);
    off += ((nfloats * sizeof(float) + 255) / 256) * 256;
    return p;
  };
  const size_t BIGN = (size_t)65536 * 384;
  float* big[3] = {alloc(BIGN), alloc(BIGN), alloc(BIGN)};
  float* mixin = alloc((size_t)512 * 898);
  float* hbuf  = alloc(512 * 128);
  float* zbuf  = alloc(512 * 128);
  float* qb    = alloc(512 * 128);
  float* kb    = alloc(512 * 128);
  float* vb    = alloc(512 * 128);
  float* qdash = alloc((size_t)4 * 512 * 256);
  float* kdash = alloc((size_t)4 * 512 * 256);
  float* diagq = alloc(512 * 4);
  float* diagk = alloc(512 * 4);
  float* qmaxb = alloc(4 * 512);
  float* kmaxb = alloc(4);
  float* kvb   = alloc(4 * 256 * 32);
  float* ksumb = alloc(4 * 256);
  float* attb  = alloc(512 * 128);
  float* t1b   = alloc((size_t)512 * 512);
  float* tfp   = alloc(128);
  float* h16   = alloc(512 * 16);

  auto cdiv = [](int a, int b) { return (a + b - 1) / b; };

  // ================= stem =================
  // x = conv1d(sequence, k15)  (M=65536, K=60, N=384)
  launch_gemm(stream, sequence, 65536, 4, 0, 4, 15, 7,
              stem_conv_k, 384, 0, nullptr, nullptr, 1.f, 0, 1.f,
              stem_conv_b, nullptr, 0, big[0], 384, 65536, 384, 60, 0);
  // x = x + conv1x1(gelu(bn(x)))
  launch_gemm(stream, big[0], 65536, 384, 0, 384, 1, 0,
              stem_res_k, 384, 0, stem_res_g, stem_res_b, BN_RS, 1, 1.f,
              stem_res_kb, big[0], 384, big[1], 384, 65536, 384, 384, 0);
  // pool logits = x @ pool_w ; then softmax-pool
  launch_gemm(stream, big[1], 65536, 384, 0, 384, 1, 0,
              stem_pool_w, 384, 0, nullptr, nullptr, 1.f, 0, 1.f,
              nullptr, nullptr, 0, big[2], 384, 65536, 384, 384, 0);
  pool_kernel<<<cdiv(32768 * 384, 256), 256, 0, stream>>>(big[1], big[2], big[0], 32768, 384);

  // ================= tower =================
  int cur = 0, curL = 32768, curC = 384;
  for (int blk = 0; blk < 6; ++blk) {
    const int f = FILT[blk];
    const int a = cur, b = (cur + 1) % 3, c = (cur + 2) % 3;
    // y = conv5(gelu(bn(x)))
    launch_gemm(stream, big[a], curL, curC, 0, curC, 5, 2,
                tk1[blk], f, 0, tg1[blk], tb1[blk], BN_RS, 1, 1.f,
                tkb1[blk], nullptr, 0, big[b], f, curL, f, 5 * curC, 0);
    // y = y + conv1x1(gelu(bn(y)))
    launch_gemm(stream, big[b], curL, f, 0, f, 1, 0,
                tk2[blk], f, 0, tg2[blk], tb2[blk], BN_RS, 1, 1.f,
                tkb2[blk], big[b], f, big[c], f, curL, f, f, 0);
    // pool logits + softmax-pool
    launch_gemm(stream, big[c], curL, f, 0, f, 1, 0,
                tpw[blk], f, 0, nullptr, nullptr, 1.f, 0, 1.f,
                nullptr, nullptr, 0, big[a], f, curL, f, f, 0);
    pool_kernel<<<cdiv((curL / 2) * f, 256), 256, 0, stream>>>(big[c], big[a], big[b], curL / 2, f);
    cur = b; curL /= 2; curC = f;
  }
  // cur: (512, 768)

  // ================= mix =================
  tf_kernel<<<1, 128, 0, stream>>>(tf_in, tf_w1, tf_b1, tf_w2, tf_b2, tfp);
  concat_kernel<<<cdiv(512 * 898, 256), 256, 0, stream>>>(big[cur], 768, tfp, 128, TSSs, exons, mixin, 512);
  launch_gemm(stream, mixin, 512, 898, 0, 898, 1, 0,
              mix_k, 128, 0, nullptr, nullptr, 1.f, 0, 1.f,
              mix_b, nullptr, 0, hbuf, 128, 512, 128, 898, 0);
  addpe_kernel<<<cdiv(512 * 128, 256), 256, 0, stream>>>(hbuf, 512, 128);

  // ================= transformer (FAVOR+) =================
  for (int L = 0; L < 4; ++L) {
    ln_kernel<<<512, 128, 0, stream>>>(hbuf, ln1g[L], ln1b[L], zbuf, 128);
    launch_gemm(stream, zbuf, 512, 128, 0, 128, 1, 0, wq[L], 128, 0,
                nullptr, nullptr, 1.f, 0, 1.f, nullptr, nullptr, 0, qb, 128, 512, 128, 128, 0);
    launch_gemm(stream, zbuf, 512, 128, 0, 128, 1, 0, wk[L], 128, 0,
                nullptr, nullptr, 1.f, 0, 1.f, nullptr, nullptr, 0, kb, 128, 512, 128, 128, 0);
    launch_gemm(stream, zbuf, 512, 128, 0, 128, 1, 0, wv[L], 128, 0,
                nullptr, nullptr, 1.f, 0, 1.f, nullptr, nullptr, 0, vb, 128, 512, 128, 128, 0);
    rotary_kernel<<<cdiv(512 * 4 * 16, 256), 256, 0, stream>>>(qb, 512, 4, 32);
    rotary_kernel<<<cdiv(512 * 4 * 16, 256), 256, 0, stream>>>(kb, 512, 4, 32);
    // dash = (x*dn) @ proj^T per head
    for (int h = 0; h < 4; ++h) {
      launch_gemm(stream, qb, 512, 128, h * 32, 32, 1, 0, proj, 32, 1,
                  nullptr, nullptr, 1.f, 0, DN, nullptr, nullptr, 0,
                  qdash + (size_t)h * 512 * 256, 256, 512, 256, 32, 0);
      launch_gemm(stream, kb, 512, 128, h * 32, 32, 1, 0, proj, 32, 1,
                  nullptr, nullptr, 1.f, 0, DN, nullptr, nullptr, 0,
                  kdash + (size_t)h * 512 * 256, 256, 512, 256, 32, 0);
    }
    diag_kernel<<<cdiv(512 * 4, 256), 256, 0, stream>>>(qb, diagq, 512, 4, DN);
    diag_kernel<<<cdiv(512 * 4, 256), 256, 0, stream>>>(kb, diagk, 512, 4, DN);
    rowmax_kernel<<<cdiv(4 * 512, 256), 256, 0, stream>>>(qdash, qmaxb, 4 * 512, 256);
    headmax_kernel<<<4, 256, 0, stream>>>(kdash, kmaxb, 512 * 256);
    phi_kernel<<<cdiv(4 * 512 * 256, 256), 256, 0, stream>>>(qdash, diagq, qmaxb, 512, 4, 256, 1, RATIO, 1e-3f);
    phi_kernel<<<cdiv(4 * 512 * 256, 256), 256, 0, stream>>>(kdash, diagk, kmaxb, 512, 4, 256, 0, RATIO, 1e-3f);
    kv_kernel<<<cdiv(4 * 256, 256), 256, 0, stream>>>(kdash, vb, kvb, ksumb, 512, 4, 256);
    // att = qp @ kv per head  (-> [512, 128] column blocks)
    for (int h = 0; h < 4; ++h) {
      launch_gemm(stream, qdash + (size_t)h * 512 * 256, 512, 256, 0, 256, 1, 0,
                  kvb + (size_t)h * 256 * 32, 32, 0,
                  nullptr, nullptr, 1.f, 0, 1.f, nullptr, nullptr, 0,
                  attb + h * 32, 128, 512, 32, 256, 0);
    }
    denomdiv_kernel<<<cdiv(512 * 4, 256), 256, 0, stream>>>(qdash, ksumb, attb, 512, 4, 256, 1e-3f);
    // h += att @ wo + wo_b
    launch_gemm(stream, attb, 512, 128, 0, 128, 1, 0, wo[L], 128, 0,
                nullptr, nullptr, 1.f, 0, 1.f, wob[L], hbuf, 128, hbuf, 128, 512, 128, 128, 0);
    // FFN
    ln_kernel<<<512, 128, 0, stream>>>(hbuf, ln2g[L], ln2b[L], zbuf, 128);
    launch_gemm(stream, zbuf, 512, 128, 0, 128, 1, 0, f1[L], 512, 0,
                nullptr, nullptr, 1.f, 0, 1.f, f1b[L], nullptr, 0, t1b, 512, 512, 512, 128, 0);
    launch_gemm(stream, t1b, 512, 512, 0, 512, 1, 0, f2[L], 128, 0,
                nullptr, nullptr, 1.f, 1, 1.f, f2b[L], hbuf, 128, hbuf, 128, 512, 128, 512, 0);
  }

  // ================= head =================
  ln_kernel<<<512, 128, 0, stream>>>(hbuf, final_ln_g, final_ln_b, zbuf, 128);
  // h16 = gelu(conv1x1(gelu(bn(z))))   (K=128 -> N=16, post-GELU fused)
  launch_gemm(stream, zbuf, 512, 128, 0, 128, 1, 0,
              fp_k, 16, 0, fp_g, fp_b, BN_RS, 1, 1.f,
              fp_kb, nullptr, 0, h16, 16, 512, 16, 128, 1);
  head_kernel<<<cdiv(512, 128), 128, 0, stream>>>(h16, head_w1, head_b1, head_w2, head_b2,
                                                  (float*)d_out, 512);
}
